// ClusteringModule_10557029613578
// MI455X (gfx1250) — matrix operand order repaired and checked by reading, not compiled, not run on hardware
//
#include <hip/hip_runtime.h>

// ---------------- problem constants ----------------
#define IMG_W 1280
#define IMG_H 720
#define OS    7
#define BN    16
#define SN    32
#define M_ROIS 512           // BN*SN
#define NCH   480
#define TT    4
#define FY    45
#define FX    80
#define DIN   23520          // NCH*OS*OS
#define NDF   1024
#define KB    735            // DIN / 32 (K-blocks per WMMA)
#define NCL   50

typedef __attribute__((ext_vector_type(16))) _Float16 v16h;
typedef __attribute__((ext_vector_type(8)))  float    v8f;

// ---------------------------------------------------------------------------
// Kernel 1: convert w_vis (f32, [NDF][DIN]) -> f16 packed in WMMA B-fragment
// layout.  B tile is 32(K) x 16(N).  Per ISA 7.12.2 (16-bit B, wave32):
//   lanes 0-15:  e0..7 = K0..7,  e8..15 = K16..23  (column N = lane)
//   lanes 16-31: e0..7 = K8..15, e8..15 = K24..31  (column N = lane-16)
// Packed addr: (((nt*KB + kb)*32 + lane)*16 + e) halves.
// ---------------------------------------------------------------------------
__global__ __launch_bounds__(256) void pack_w_kernel(const float* __restrict__ w_vis,
                                                     _Float16* __restrict__ Bpk) {
    int t = blockIdx.x * 256 + threadIdx.x;
    if (t >= NDF * DIN) return;
    int n  = t / DIN;
    int k  = t - n * DIN;
    int nt = n >> 4, col = n & 15;
    int kb = k >> 5, kk  = k & 31;
    int lane = col + (((kk >> 3) & 1) << 4);
    int e    = (kk & 7) + ((kk >> 4) << 3);
    Bpk[(((size_t)nt * KB + kb) * 32 + lane) * 16 + e] = (_Float16)w_vis[t];
}

// ---------------------------------------------------------------------------
// Kernel 2: RoI-align (sampling_ratio=1, aligned=True) on last time slice,
// writing pooled values f32->f16 directly into WMMA A-fragment layout.
// A tile is 16(M) x 32(K), 16-bit (ISA 7.12.2):
//   lanes 0-15:  e0..7 = K0..7,  e8..15 = K16..23   (row M = lane)
//   lanes 16-31: e0..7 = K8..15, e8..15 = K24..31   (row M = lane-16)
// Packed addr: (((mt*KB + kb)*32 + lane)*16 + e) halves.
// One block per roi.
// ---------------------------------------------------------------------------
__global__ __launch_bounds__(256) void roi_pack_kernel(const float* __restrict__ zv,
                                                       const float* __restrict__ bboxs,
                                                       _Float16* __restrict__ Apk) {
    __shared__ float W00[49], W01[49], W10[49], W11[49];
    __shared__ int   I00[49], I01[49], I10[49], I11[49];

    int m   = blockIdx.x;          // roi index 0..511
    int b   = m >> 5;              // batch = m / SN
    int tid = threadIdx.x;

    if (tid < 49) {
        const float* bb = bboxs + (size_t)m * 4;
        const float sx = (float)FX / (float)IMG_W;
        const float sy = (float)FY / (float)IMG_H;
        float v0 = bb[0] * sx;  v0 = (v0 == v0) ? v0 : 0.0f;   // nan_to_num
        float v1 = bb[1] * sy;  v1 = (v1 == v1) ? v1 : 0.0f;
        float v2 = bb[2] * sx;  v2 = (v2 == v2) ? v2 : 0.0f;
        float v3 = bb[3] * sy;  v3 = (v3 == v3) ? v3 : 0.0f;
        float x1 = v0 - 0.5f, y1 = v1 - 0.5f;
        float x2 = v2 - 0.5f, y2 = v3 - 0.5f;
        float bwd = (x2 - x1) * (1.0f / (float)OS);
        float bhd = (y2 - y1) * (1.0f / (float)OS);
        int iy = tid / OS, ix = tid - iy * OS;
        float y = y1 + ((float)iy + 0.5f) * bhd;
        float x = x1 + ((float)ix + 0.5f) * bwd;
        float vf = (y > -1.0f && y < (float)FY && x > -1.0f && x < (float)FX) ? 1.0f : 0.0f;
        float yc = fminf(fmaxf(y, 0.0f), (float)(FY - 1));
        float xc = fminf(fmaxf(x, 0.0f), (float)(FX - 1));
        int yl = (int)floorf(yc), xl = (int)floorf(xc);
        int yh = min(yl + 1, FY - 1), xh = min(xl + 1, FX - 1);
        float ly = yc - (float)yl, lx = xc - (float)xl;
        float hy = 1.0f - ly, hx = 1.0f - lx;
        W00[tid] = hy * hx * vf;  W01[tid] = hy * lx * vf;
        W10[tid] = ly * hx * vf;  W11[tid] = ly * lx * vf;
        I00[tid] = yl * FX + xl;  I01[tid] = yl * FX + xh;
        I10[tid] = yh * FX + xl;  I11[tid] = yh * FX + xh;
    }
    __syncthreads();

    int mt = m >> 4, lm = m & 15;
    // z_vis element (b,c,TT-1,y,x): ((b*NCH + c)*TT + TT-1)*FY*FX + y*FX + x
    const float* featBase = zv + ((size_t)b * NCH * TT + (TT - 1)) * (FY * FX);

    for (int k = tid; k < DIN; k += 256) {
        int c = k / 49, p = k - c * 49;   // k = c*49 + iy*7 + ix  (matches [N,C,7,7] flatten)
        const float* fp = featBase + (size_t)c * (TT * FY * FX);
        float val = fp[I00[p]] * W00[p] + fp[I01[p]] * W01[p]
                  + fp[I10[p]] * W10[p] + fp[I11[p]] * W11[p];
        int kb = k >> 5, kk = k & 31;
        int hi = kk >> 3;
        int lane = lm + ((hi & 1) << 4);
        int e    = (kk & 7) + ((hi >> 1) << 3);
        Apk[(((size_t)mt * KB + kb) * 32 + lane) * 16 + e] = (_Float16)val;
    }
}

// ---------------------------------------------------------------------------
// Kernel 3: GEMM z = A(512x23520) * B(23520x1024) via v_wmma_f32_16x16x32_f16,
// fused epilogue: + b_vis[n] + norms[m]*w_spc[n] + b_spc[n].
// Block = 256 threads = 8 waves.  grid = (32 M-tiles, 4 N-supertiles).
// Each wave: one 16-row M tile x two 16-col N tiles, K loop of 735.
// No explicit prefetch: operands are L2-resident (72 MB packed vs 192 MB L2)
// and the compiler's load pipelining (loadcnt<=4) covers L2 latency; the
// prefetch address math was costing SALU/VMEM issue slots in the hot loop.
// ---------------------------------------------------------------------------
__global__ __launch_bounds__(256) void gemm_wmma_kernel(const v16h* __restrict__ Apk,
                                                        const v16h* __restrict__ Bpk,
                                                        const float* __restrict__ b_vis,
                                                        const float* __restrict__ w_spc,
                                                        const float* __restrict__ b_spc,
                                                        const float* __restrict__ norms,
                                                        float* __restrict__ z_out) {
    int lane = threadIdx.x & 31;
    int wave = threadIdx.x >> 5;
    int mt   = blockIdx.x;                       // 0..31
    int nt0  = blockIdx.y * 16 + wave * 2;       // 0..62 (even)

    const v16h* aP  = Apk + (size_t)mt * KB * 32 + lane;
    const v16h* bP0 = Bpk + (size_t)nt0 * KB * 32 + lane;
    const v16h* bP1 = Bpk + ((size_t)nt0 + 1) * KB * 32 + lane;

    v8f acc0 = {};
    v8f acc1 = {};

    for (int kb = 0; kb < KB; ++kb) {
        v16h a  = aP [(size_t)kb * 32];
        v16h b0 = bP0[(size_t)kb * 32];
        v16h b1 = bP1[(size_t)kb * 32];
        acc0 = __builtin_amdgcn_wmma_f32_16x16x32_f16(false, a, false, b0,
                                                      (short)0, acc0, false, false);
        acc1 = __builtin_amdgcn_wmma_f32_16x16x32_f16(false, a, false, b1,
                                                      (short)0, acc1, false, false);
    }

    // C/D layout (ISA 7.12.2): VGPR r -> (M = r, N = lane) lanes 0-15,
    //                                    (M = r+8, N = lane-16) lanes 16-31.
    int col   = lane & 15;
    int rh    = lane >> 4;
    int mBase = mt * 16;
    {
        int n = nt0 * 16 + col;
        float add = b_vis[n] + b_spc[n];
        float ws  = w_spc[n];
#pragma unroll
        for (int r = 0; r < 8; ++r) {
            int mm = mBase + r + 8 * rh;
            z_out[(size_t)mm * NDF + n] = acc0[r] + add + norms[mm] * ws;
        }
    }
    {
        int n = (nt0 + 1) * 16 + col;
        float add = b_vis[n] + b_spc[n];
        float ws  = w_spc[n];
#pragma unroll
        for (int r = 0; r < 8; ++r) {
            int mm = mBase + r + 8 * rh;
            z_out[(size_t)mm * NDF + n] = acc1[r] + add + norms[mm] * ws;
        }
    }
}

// ---------------------------------------------------------------------------
// Kernel 4: Student-t cluster assignment.  One block per roi row.
// dist_j = ||z - centroid_j||; s_j = (1+dist_j)^-1 normalized; c = argmax s.
// ---------------------------------------------------------------------------
__global__ __launch_bounds__(256) void cluster_kernel(const float* __restrict__ z,
                                                      const float* __restrict__ centroids,
                                                      float* __restrict__ s_out,
                                                      float* __restrict__ c_out) {
    __shared__ float zsh[NDF];
    __shared__ float red[8];
    __shared__ float stmp[NCL];

    int m = blockIdx.x;
    for (int i = threadIdx.x; i < NDF; i += 256) zsh[i] = z[(size_t)m * NDF + i];
    __syncthreads();

    for (int j = 0; j < NCL; ++j) {
        float p = 0.0f;
        const float* cj = centroids + (size_t)j * NDF;
        for (int i = threadIdx.x; i < NDF; i += 256) {
            float d = zsh[i] - cj[i];
            p += d * d;
        }
        for (int off = 16; off > 0; off >>= 1) p += __shfl_down(p, off, 32);
        if ((threadIdx.x & 31) == 0) red[threadIdx.x >> 5] = p;
        __syncthreads();
        if (threadIdx.x == 0) {
            float tot = 0.0f;
#pragma unroll
            for (int w = 0; w < 8; ++w) tot += red[w];
            stmp[j] = 1.0f / (1.0f + sqrtf(tot));   // ALPHA = 1
        }
        __syncthreads();
    }

    if (threadIdx.x == 0) {
        float sum = 0.0f;
        for (int j = 0; j < NCL; ++j) sum += stmp[j];
        float inv = 1.0f / sum;
        float best = -1.0f;
        int bi = 0;
        for (int j = 0; j < NCL; ++j) {
            float v = stmp[j] * inv;
            s_out[(size_t)m * NCL + j] = v;
            if (v > best) { best = v; bi = j; }
        }
        c_out[m] = (float)bi;
    }
}

// ---------------------------------------------------------------------------
extern "C" void kernel_launch(void* const* d_in, const int* in_sizes, int n_in,
                              void* d_out, int out_size, void* d_ws, size_t ws_size,
                              hipStream_t stream) {
    const float* z_vis     = (const float*)d_in[0];
    const float* bboxs     = (const float*)d_in[1];
    const float* norms     = (const float*)d_in[2];
    const float* w_vis     = (const float*)d_in[3];
    const float* b_vis     = (const float*)d_in[4];
    const float* w_spc     = (const float*)d_in[5];
    const float* b_spc     = (const float*)d_in[6];
    const float* centroids = (const float*)d_in[7];

    float* out   = (float*)d_out;
    float* z_out = out;                                   // [512][1024]
    float* s_out = out + (size_t)M_ROIS * NDF;            // [512][50]
    float* c_out = s_out + (size_t)M_ROIS * NCL;          // [512]

    // Workspace: Apk 512*23520*2B = 24.1 MB, Bpk 1024*23520*2B = 48.2 MB
    _Float16* Apk = (_Float16*)d_ws;
    _Float16* Bpk = Apk + (size_t)M_ROIS * DIN;

    // 1) pack weights -> f16 B fragments
    int wElems  = NDF * DIN;
    pack_w_kernel<<<(wElems + 255) / 256, 256, 0, stream>>>(w_vis, Bpk);

    // 2) roi-align -> f16 A fragments
    roi_pack_kernel<<<M_ROIS, 256, 0, stream>>>(z_vis, bboxs, Apk);

    // 3) WMMA GEMM + fused bias / spatial term
    dim3 grid(M_ROIS / 16, NDF / 256);                    // (32, 4)
    gemm_wmma_kernel<<<grid, 256, 0, stream>>>((const v16h*)Apk, (const v16h*)Bpk,
                                               b_vis, w_spc, b_spc, norms, z_out);

    // 4) cluster distances / soft-assignment / argmax
    cluster_kernel<<<M_ROIS, 256, 0, stream>>>(z_out, centroids, s_out, c_out);
}
